// Dcnv2_6279242186997
// MI455X (gfx1250) — matrix operand order repaired and checked by reading, not compiled, not run on hardware
//
#include <hip/hip_runtime.h>

typedef _Float16 v16h __attribute__((ext_vector_type(16)));
typedef _Float16 v4h  __attribute__((ext_vector_type(4)));
typedef float    v8f  __attribute__((ext_vector_type(8)));
typedef float    v4f  __attribute__((ext_vector_type(4)));

#define BB   4
#define CIN  64
#define COUT 64
#define HH   128
#define WW   128
#define HW   (128*128)

// ---------------------------------------------------------------------------
// Kernel 0: NCHW -> NHWC transpose of x via LDS tile (bandwidth-trivial).
// ---------------------------------------------------------------------------
__global__ __launch_bounds__(256) void nchw_to_nhwc(
    const float* __restrict__ x, float* __restrict__ xt)
{
    __shared__ float tile[64][65];
    const int p0 = blockIdx.x * 64;
    const int b  = p0 >> 14;
    const int sp = p0 & (HW - 1);
#pragma unroll
    for (int i = 0; i < 16; ++i) {
        int e = threadIdx.x + 256 * i;
        int c = e >> 6, q = e & 63;
        tile[q][c] = x[(b * CIN + c) * HW + sp + q];
    }
    __syncthreads();
#pragma unroll
    for (int i = 0; i < 16; ++i) {
        int e = threadIdx.x + 256 * i;
        int q = e >> 6, c = e & 63;
        xt[(size_t)(p0 + q) * CIN + c] = tile[q][c];
    }
}

// Stage A-matrix (reordered K = kk*64 + c) into LDS in the exact per-lane
// order each lane loads as its A operand, 4 K-contiguous halves per store.
// In the ISA layout, VGPR v holds K = (v<4 ? 2v : 16+2(v-4)) + 8*(lane>=16),
// but here element e holds K = 16*(lane>=16) + e, so group g: K-base = 4g + 16*(lane>=16).
template <int MT>
__device__ __forceinline__ void stage_A(
    _Float16 (*lA)[18][32][16], const float* __restrict__ wsrc, bool padM)
{
    for (int t = threadIdx.x; t < MT * 18 * 32 * 4; t += 256) {
        int g  = t & 3;
        int ln = (t >> 2) & 31;
        int sl = (t >> 7) % 18;
        int mt = t / (18 * 32 * 4);
        int kb = 4 * g + ((ln >> 4) << 4);
        int M  = mt * 16 + (ln & 15);
        v4h val;
#pragma unroll
        for (int j = 0; j < 4; ++j) {
            int Kidx = sl * 32 + kb + j;
            int kk = Kidx >> 6;          // tap
            int c  = Kidx & 63;          // channel
            val[j] = (padM && M >= 18) ? (_Float16)0.f
                                       : (_Float16)wsrc[(M * CIN + c) * 9 + kk];
        }
        *(v4h*)&lA[mt][sl][ln][4 * g] = val;
    }
}

// ---------------------------------------------------------------------------
// Kernel 1: offset conv (18 out ch -> M padded to 32) as WMMA GEMM.
// ---------------------------------------------------------------------------
__global__ __launch_bounds__(256) void dcn_offset_wmma(
    const float* __restrict__ xt, const float* __restrict__ w,
    float* __restrict__ off)
{
    __shared__ __align__(32) _Float16 lA[2][18][32][16];   // 36 KB

    const int lane = threadIdx.x & 31;
    const int wave = threadIdx.x >> 5;
    const int tile = blockIdx.x * 8 + wave;
    const int p    = tile * 16 + (lane & 15);
    const int px   = p & 127;
    const int py   = (p >> 7) & 127;
    const int b    = p >> 14;

    stage_A<2>(lA, w, /*padM=*/true);
    __syncthreads();

    const float* xtb  = xt + (size_t)b * HW * CIN;
    const int    coff = (lane & 16) ? 16 : 0;
    v8f acc[2] = {};

    v16h afr[2];
#pragma unroll
    for (int mt = 0; mt < 2; ++mt)
        afr[mt] = *(const v16h*)&lA[mt][0][lane][0];

    for (int kk = 0; kk < 9; ++kk) {
        int ki = kk / 3, kj = kk % 3;
        int y  = py + ki - 1, xx = px + kj - 1;
        float msk = (((unsigned)y < 128u) && ((unsigned)xx < 128u)) ? 1.f : 0.f;
        int yc = min(max(y, 0), 127), xc = min(max(xx, 0), 127);
        int oc = (yc * WW + xc) << 6;
#pragma unroll
        for (int hf = 0; hf < 2; ++hf) {
            int s = kk * 2 + hf;
            const float* src = xtb + oc + ((hf << 5) + coff);
            v16h bfrag;
#pragma unroll
            for (int g = 0; g < 4; ++g) {
                v4f v = msk * (*(const v4f*)(src + 4 * g));
                bfrag[4 * g + 0] = (_Float16)v.x;
                bfrag[4 * g + 1] = (_Float16)v.y;
                bfrag[4 * g + 2] = (_Float16)v.z;
                bfrag[4 * g + 3] = (_Float16)v.w;
            }
            int sn = (s < 17) ? s + 1 : 17;      // branchless prefetch index
            v16h anx[2];
#pragma unroll
            for (int mt = 0; mt < 2; ++mt)
                anx[mt] = *(const v16h*)&lA[mt][sn][lane][0];
#pragma unroll
            for (int mt = 0; mt < 2; ++mt)
                acc[mt] = __builtin_amdgcn_wmma_f32_16x16x32_f16(
                    false, afr[mt], false, bfrag, (short)0, acc[mt], false, false);
#pragma unroll
            for (int mt = 0; mt < 2; ++mt) afr[mt] = anx[mt];
        }
    }

    const int hi8 = (lane >> 4) << 3;
#pragma unroll
    for (int mt = 0; mt < 2; ++mt)
#pragma unroll
        for (int r = 0; r < 8; ++r) {
            int M = mt * 16 + hi8 + r;
            if (M < 18)
                off[((b * 18 + M) * HH + py) * WW + px] = acc[mt][r];
        }
}

// ---------------------------------------------------------------------------
// Kernel 2: deformable bilinear sampling + 64x576 WMMA contraction.
// All 18 K-steps of A resident in 72 KB LDS; pipelined A loads; branch-free
// inner body (coords once per tap).
// ---------------------------------------------------------------------------
__global__ __launch_bounds__(256) void dcn_deform_wmma(
    const float* __restrict__ xt, const float* __restrict__ wd,
    const float* __restrict__ off, float* __restrict__ out)
{
    __shared__ __align__(32) _Float16 lA[4][18][32][16];   // 72 KB

    const int lane = threadIdx.x & 31;
    const int wave = threadIdx.x >> 5;
    const int tile = blockIdx.x * 8 + wave;
    const int p    = tile * 16 + (lane & 15);
    const int px   = p & 127;
    const int py   = (p >> 7) & 127;
    const int b    = p >> 14;

    stage_A<4>(lA, wd, /*padM=*/false);
    __syncthreads();

    const float* xtb  = xt + (size_t)b * HW * CIN;
    const int    coff = (lane & 16) ? 16 : 0;
    v8f acc[4] = {};

    v16h afr[4];
#pragma unroll
    for (int mt = 0; mt < 4; ++mt)
        afr[mt] = *(const v16h*)&lA[mt][0][lane][0];

    for (int kk = 0; kk < 9; ++kk) {
        // ---- bilinear tap: clamped corner offsets + border-masked weights
        int   ki  = kk / 3, kj = kk % 3;
        float dyv = off[((b * 18 + 2 * kk) * HH + py) * WW + px];
        float dxv = off[((b * 18 + 2 * kk + 1) * HH + py) * WW + px];
        float ys  = (float)(py - 1 + ki) + dyv;
        float xs  = (float)(px - 1 + kj) + dxv;
        float y0f = floorf(ys), x0f = floorf(xs);
        float wy  = ys - y0f,   wx  = xs - x0f;
        int y0 = (int)y0f, x0 = (int)x0f;
        int y1 = y0 + 1,   x1 = x0 + 1;
        float vy0 = ((unsigned)y0 < 128u) ? 1.f : 0.f;
        float vy1 = ((unsigned)y1 < 128u) ? 1.f : 0.f;
        float vx0 = ((unsigned)x0 < 128u) ? 1.f : 0.f;
        float vx1 = ((unsigned)x1 < 128u) ? 1.f : 0.f;
        int yc0 = min(max(y0, 0), 127), yc1 = min(max(y1, 0), 127);
        int xc0 = min(max(x0, 0), 127), xc1 = min(max(x1, 0), 127);
        float w00 = (1.f - wy) * (1.f - wx) * vy0 * vx0;
        float w01 = (1.f - wy) * wx         * vy0 * vx1;
        float w10 = wy         * (1.f - wx) * vy1 * vx0;
        float w11 = wy         * wx         * vy1 * vx1;
        int o00 = (yc0 * WW + xc0) << 6,  o01 = (yc0 * WW + xc1) << 6;
        int o10 = (yc1 * WW + xc0) << 6,  o11 = (yc1 * WW + xc1) << 6;

#pragma unroll
        for (int hf = 0; hf < 2; ++hf) {
            int s = kk * 2 + hf;
            int cstart = (hf << 5) + coff;
            v16h bfrag;
#pragma unroll
            for (int g = 0; g < 4; ++g) {
                int cg = cstart + 4 * g;
                v4f a  = *(const v4f*)(xtb + o00 + cg);
                v4f bb = *(const v4f*)(xtb + o01 + cg);
                v4f c2 = *(const v4f*)(xtb + o10 + cg);
                v4f d2 = *(const v4f*)(xtb + o11 + cg);
                v4f sv = w00 * a + w01 * bb + w10 * c2 + w11 * d2;
                bfrag[4 * g + 0] = (_Float16)sv.x;
                bfrag[4 * g + 1] = (_Float16)sv.y;
                bfrag[4 * g + 2] = (_Float16)sv.z;
                bfrag[4 * g + 3] = (_Float16)sv.w;
            }
            int sn = (s < 17) ? s + 1 : 17;      // branchless prefetch index
            v16h anx[4];
#pragma unroll
            for (int mt = 0; mt < 4; ++mt)
                anx[mt] = *(const v16h*)&lA[mt][sn][lane][0];
#pragma unroll
            for (int mt = 0; mt < 4; ++mt)
                acc[mt] = __builtin_amdgcn_wmma_f32_16x16x32_f16(
                    false, afr[mt], false, bfrag, (short)0, acc[mt], false, false);
#pragma unroll
            for (int mt = 0; mt < 4; ++mt) afr[mt] = anx[mt];
        }
    }

    const int hi8 = (lane >> 4) << 3;
#pragma unroll
    for (int mt = 0; mt < 4; ++mt)
#pragma unroll
        for (int r = 0; r < 8; ++r) {
            int M = mt * 16 + hi8 + r;
            out[(b * COUT + M) * HW + py * WW + px] = acc[mt][r];
        }
}

// ---------------------------------------------------------------------------
extern "C" void kernel_launch(void* const* d_in, const int* in_sizes, int n_in,
                              void* d_out, int out_size, void* d_ws, size_t ws_size,
                              hipStream_t stream)
{
    const float* x        = (const float*)d_in[0];   // (4, 64, 128, 128)
    const float* w_offset = (const float*)d_in[1];   // (18, 64, 3, 3)
    const float* w_deform = (const float*)d_in[2];   // (64, 64, 3, 3)
    float*       out      = (float*)d_out;           // (4, 64, 128, 128)

    float* off = (float*)d_ws;                       // (4,18,128,128) 4.7 MB
    float* xt  = off + (size_t)BB * 18 * HW;         // NHWC x, 16.8 MB

    nchw_to_nhwc   <<<(BB * HW) / 64, 256, 0, stream>>>(x, xt);
    dcn_offset_wmma<<<(BB * HW / 16) / 8, 256, 0, stream>>>(xt, w_offset, off);
    dcn_deform_wmma<<<(BB * HW / 16) / 8, 256, 0, stream>>>(xt, w_deform, off, out);
}
